// Pronouncer_5875515261015
// MI455X (gfx1250) — compile-verified, matrix operand order checked by reading, not run on hardware
//
#include <hip/hip_runtime.h>

#define D_   320
#define H_   256
#define C_   128
#define J_   512
#define DC_  448
#define U_   50
#define MROWS 64   // rows per block

// f16 weight regions inside d_ws (element offsets)
#define NP_  (C_ * J_)
#define NI_  (H_ * DC_)
#define NH_  (H_ * H_)
#define NO_  (2 * D_ * H_)
#define OFF_P 0
#define OFF_I (OFF_P + NP_)
#define OFF_H (OFF_I + NI_)
#define OFF_O (OFF_H + NH_)
#define NW_TOT (OFF_O + NO_)      // 409600 halfs = 819200 bytes

typedef __attribute__((ext_vector_type(16))) _Float16 v16h;
typedef __attribute__((ext_vector_type(8)))  _Float16 v8h;
typedef __attribute__((ext_vector_type(8)))  float    v8f;

__device__ __forceinline__ v8f wmma_f16(v16h a, v16h b, v8f c) {
    return __builtin_amdgcn_wmma_f32_16x16x32_f16(
        false, a, false, b, (short)0, c, false, false);
}

// 16x32 fragment from an f16 row (LDS or global): lane holds M(or N)=lane%16;
// half i -> K = kbase + (i%8) + 16*(i/8) + 8*hi. Two 16B vector loads.
__device__ __forceinline__ v16h frag16(const _Float16* row, int kbase, int hi) {
    const _Float16* p = row + kbase + 8 * hi;
    v8h a = *(const v8h*)(p);
    v8h b = *(const v8h*)(p + 16);
    v16h f;
#pragma unroll
    for (int i = 0; i < 8; ++i) { f[i] = a[i]; f[8 + i] = b[i]; }
    return f;
}

__device__ __forceinline__ float dswish(float x) {
    return x / (1.0f + __expf(1.0f - x));  // x * sigmoid(x-1)
}

// ---- prep: convert weights f32->f16 with MADE masks baked in ----
__global__ __launch_bounds__(256) void prep_weights(
    const float* __restrict__ Wp, const float* __restrict__ Wi,
    const float* __restrict__ Wh, const float* __restrict__ Wo,
    _Float16* __restrict__ w16) {
    const int i = blockIdx.x * 256 + threadIdx.x;
    if (i >= NW_TOT) return;
    float w;
    if (i < OFF_I) {                       // W_proj: no mask
        w = Wp[i];
    } else if (i < OFF_H) {                // W_in: m0[h,d] = (h >= d) on first 320 cols
        const int j = i - OFF_I;
        const int oc = j / DC_, k = j - oc * DC_;
        w = Wi[j];
        if (k < D_ && oc < k) w = 0.0f;
    } else if (i < OFF_O) {                // W_h: m1[g,h] = (g >= h)
        const int j = i - OFF_H;
        const int oc = j / H_, k = j - oc * H_;
        w = (oc >= k) ? Wh[j] : 0.0f;
    } else {                               // W_out: m2[o,h] = ((o%320) > h)
        const int j = i - OFF_O;
        const int oc = j / H_, k = j - oc * H_;
        w = ((oc % D_) > k) ? Wo[j] : 0.0f;
    }
    w16[i] = (_Float16)w;
}

__global__ __launch_bounds__(256) void made_flow_fused(
    const float* __restrict__ joint, const float* __restrict__ xtg,
    const _Float16* __restrict__ w16,
    const float* __restrict__ bp, const float* __restrict__ bi,
    const float* __restrict__ bh, const float* __restrict__ bo,
    float* __restrict__ out) {
    __shared__ __align__(16) _Float16 xt_s[MROWS * D_];    // 40 KB (persistent)
    __shared__ __align__(16) _Float16 cond_s[MROWS * C_];  // 16 KB
    __shared__ __align__(16) _Float16 pool_s[MROWS * J_];  // 64 KB: joint, then h1|h2
    __shared__ float logp_s[MROWS];

    _Float16* joint_s = pool_s;
    _Float16* h1_s    = pool_s;
    _Float16* h2_s    = pool_s + MROWS * H_;

    const _Float16* Wp16 = w16 + OFF_P;
    const _Float16* Wi16 = w16 + OFF_I;
    const _Float16* Wh16 = w16 + OFF_H;
    const _Float16* Wo16 = w16 + OFF_O;

    const int tid  = threadIdx.x;
    const int wave = tid >> 5;
    const int lane = tid & 31;
    const int hi   = lane >> 4;
    const int l16  = lane & 15;
    const int mrow = (wave & 3) * 16;  // wave's 16-row M subtile
    const int nsel = wave >> 2;        // N-range half (0/1)
    const int row0 = blockIdx.x * MROWS;

    // ---- stage 0: stage xt and joint tile into LDS (f16); zero logp ----
    for (int idx = tid; idx < MROWS * D_; idx += 256) {
        const int r = idx / D_, d = idx - r * D_;
        const int xr = (row0 + r) / U_;
        xt_s[idx] = (_Float16)xtg[(size_t)xr * D_ + d];
    }
    {
        const float* jbase = joint + (size_t)row0 * J_;
        for (int idx = tid; idx < MROWS * J_; idx += 256)
            joint_s[idx] = (_Float16)jbase[idx];
    }
    if (tid < MROWS) logp_s[tid] = 0.0f;
    __syncthreads();

    // ---- stage 1: cond = joint x Wp^T + bp  (K=512 -> N=128) ----
    // K outer, 4 N-tile accumulators live: A fragment loaded once per K slice.
    {
        const _Float16* arow = joint_s + (mrow + l16) * J_;
        v8f acc[4];
#pragma unroll
        for (int t = 0; t < 4; ++t) acc[t] = (v8f){};
        for (int k = 0; k < J_; k += 32) {
            const v16h a = frag16(arow, k, hi);
#pragma unroll
            for (int t = 0; t < 4; ++t) {
                const int oc = nsel * 64 + t * 16 + l16;
                acc[t] = wmma_f16(a, frag16(Wp16 + (size_t)oc * J_, k, hi), acc[t]);
            }
        }
#pragma unroll
        for (int t = 0; t < 4; ++t) {
            const int oc = nsel * 64 + t * 16 + l16;
            const float bias = bp[oc];
#pragma unroll
            for (int r = 0; r < 8; ++r) {
                const int M = mrow + r + 8 * hi;
                cond_s[M * C_ + oc] = (_Float16)(acc[t][r] + bias);
            }
        }
    }
    __syncthreads();  // cond ready; joint_s dead -> pool becomes h1/h2

    // ---- stage 2: h1 = dswish([xt|cond] x Wi16^T + bi)  (K=448 -> N=256) ----
    {
        const _Float16* xrow = xt_s   + (mrow + l16) * D_;
        const _Float16* crow = cond_s + (mrow + l16) * C_;
        v8f acc[8];
#pragma unroll
        for (int t = 0; t < 8; ++t) acc[t] = (v8f){};
        for (int k = 0; k < D_; k += 32) {
            const v16h a = frag16(xrow, k, hi);
#pragma unroll
            for (int t = 0; t < 8; ++t) {
                const int oc = nsel * 128 + t * 16 + l16;
                acc[t] = wmma_f16(a, frag16(Wi16 + (size_t)oc * DC_, k, hi), acc[t]);
            }
        }
        for (int k = 0; k < C_; k += 32) {
            const v16h a = frag16(crow, k, hi);
#pragma unroll
            for (int t = 0; t < 8; ++t) {
                const int oc = nsel * 128 + t * 16 + l16;
                acc[t] = wmma_f16(a, frag16(Wi16 + (size_t)oc * DC_, D_ + k, hi), acc[t]);
            }
        }
#pragma unroll
        for (int t = 0; t < 8; ++t) {
            const int oc = nsel * 128 + t * 16 + l16;
            const float bias = bi[oc];
#pragma unroll
            for (int r = 0; r < 8; ++r) {
                const int M = mrow + r + 8 * hi;
                h1_s[M * H_ + oc] = (_Float16)dswish(acc[t][r] + bias);
            }
        }
    }
    __syncthreads();

    // ---- stage 3: h2 = dswish(h1 x Wh16^T + bh)  (K=256 -> N=256) ----
    {
        const _Float16* hrow = h1_s + (mrow + l16) * H_;
        v8f acc[8];
#pragma unroll
        for (int t = 0; t < 8; ++t) acc[t] = (v8f){};
        for (int k = 0; k < H_; k += 32) {
            const v16h a = frag16(hrow, k, hi);
#pragma unroll
            for (int t = 0; t < 8; ++t) {
                const int oc = nsel * 128 + t * 16 + l16;
                acc[t] = wmma_f16(a, frag16(Wh16 + (size_t)oc * H_, k, hi), acc[t]);
            }
        }
#pragma unroll
        for (int t = 0; t < 8; ++t) {
            const int oc = nsel * 128 + t * 16 + l16;
            const float bias = bh[oc];
#pragma unroll
            for (int r = 0; r < 8; ++r) {
                const int M = mrow + r + 8 * hi;
                h2_s[M * H_ + oc] = (_Float16)dswish(acc[t][r] + bias);
            }
        }
    }
    __syncthreads();

    // ---- stage 4: (m, loga) pairs + fused Gaussian log-prob, 2 passes of 5 ----
    {
        const _Float16* hrow = h2_s + (mrow + l16) * H_;
        float plogp[8];
#pragma unroll
        for (int r = 0; r < 8; ++r) plogp[r] = 0.0f;

#pragma unroll 1
        for (int p = 0; p < 2; ++p) {
            v8f am[5], al[5];
#pragma unroll
            for (int j = 0; j < 5; ++j) { am[j] = (v8f){}; al[j] = (v8f){}; }
            for (int k = 0; k < H_; k += 32) {
                const v16h a = frag16(hrow, k, hi);
#pragma unroll
                for (int j = 0; j < 5; ++j) {
                    const int d = nsel * 160 + (p * 5 + j) * 16 + l16;  // = o % 320
                    am[j] = wmma_f16(a, frag16(Wo16 + (size_t)d * H_, k, hi), am[j]);
                    al[j] = wmma_f16(a, frag16(Wo16 + (size_t)(d + D_) * H_, k, hi), al[j]);
                }
            }
#pragma unroll
            for (int j = 0; j < 5; ++j) {
                const int d = nsel * 160 + (p * 5 + j) * 16 + l16;
                const float bm = bo[d], bl = bo[d + D_];
#pragma unroll
                for (int r = 0; r < 8; ++r) {
                    const int M = mrow + r + 8 * hi;
                    const float mval = am[j][r] + bm;
                    const float la   = al[j][r] + bl;
                    const float x = (float)xt_s[M * D_ + d];
                    const float u = (x - mval) * __expf(-la);
                    plogp[r] += -0.5f * u * u - la;
                }
            }
        }
#pragma unroll
        for (int r = 0; r < 8; ++r)
            atomicAdd(&logp_s[mrow + r + 8 * hi], plogp[r]);   // ds_add_f32
    }
    __syncthreads();

    // ---- stage 5: logp - D * 0.5*log(2*pi) ----
    if (tid < MROWS)
        out[row0 + tid] = logp_s[tid] - 294.06033f;  // 320 * 0.9189385332
}

extern "C" void kernel_launch(void* const* d_in, const int* in_sizes, int n_in,
                              void* d_out, int out_size, void* d_ws, size_t ws_size,
                              hipStream_t stream) {
    (void)in_sizes; (void)n_in; (void)ws_size;
    const float* joint = (const float*)d_in[0];
    const float* xtg   = (const float*)d_in[1];
    const float* Wp    = (const float*)d_in[2];
    const float* bp    = (const float*)d_in[3];
    const float* Wi    = (const float*)d_in[4];
    const float* bi    = (const float*)d_in[5];
    const float* Wh    = (const float*)d_in[6];
    const float* bh    = (const float*)d_in[7];
    const float* Wo    = (const float*)d_in[8];
    const float* bo    = (const float*)d_in[9];
    float* out = (float*)d_out;
    _Float16* w16 = (_Float16*)d_ws;  // needs 819200 bytes

    prep_weights<<<(NW_TOT + 255) / 256, 256, 0, stream>>>(Wp, Wi, Wh, Wo, w16);

    const int blocks = out_size / MROWS;  // 80000 / 64 = 1250
    made_flow_fused<<<blocks, 256, 0, stream>>>(joint, xtg, w16, bp, bi, bh, bo, out);
}
